// Embedding_84988812853831
// MI455X (gfx1250) — matrix operand order repaired and checked by reading, not compiled, not run on hardware
//
#include <hip/hip_runtime.h>
#include <hip/hip_bf16.h>

// ---------------------------------------------------------------------------
// Embedding pipeline for MI455X (gfx1250), wave32.
// - bf16 WMMA (v_wmma_f32_16x16x32_bf16) with f32 accumulation.
// - Tile staging via Tensor Data Mover (tensor_load_to_lds) + s_wait_tensorcnt,
//   double-buffered LDS; TDM pad feature generates the 96B padded row stride.
// - Weights pre-transposed to [N][K] so both A and B tiles are plain 2D tiles.
// ---------------------------------------------------------------------------

typedef __bf16 bf16;
typedef __attribute__((ext_vector_type(16))) __bf16 v16bf;
typedef __attribute__((ext_vector_type(8)))  __bf16 v8bf;
typedef __attribute__((ext_vector_type(8)))  float  v8f;
typedef __attribute__((ext_vector_type(4)))  unsigned int u32x4;
typedef __attribute__((ext_vector_type(8)))  int  i32x8;
typedef __attribute__((ext_vector_type(4)))  int  i32x4;

__device__ inline bf16 f2bf(float f) {
    unsigned u = __builtin_bit_cast(unsigned, f);
    unsigned r = u + 0x7FFFu + ((u >> 16) & 1u);   // round-to-nearest-even
    unsigned short h = (unsigned short)(r >> 16);
    return __builtin_bit_cast(bf16, h);
}

__device__ inline float gelu_exact(float x) {
    return 0.5f * x * (1.0f + erff(x * 0.70710678118654752f));
}

// ---------------- constants -------------------------------------------------
#define MROWS 32768      // B*T = 64*512
#define DIM 512
#define K1 160           // 132 packed features padded to multiple of 32
#define H3 1536          // 3 * DIM hidden (block-diagonal first layer)

#define TILE_M 64
#define TILE_N 128
#define TILE_K 32
#define LDS_RS 48        // padded LDS row stride (bf16): 96B (16B multiple)

// ---------------- preprocess: normalize + pack to bf16 ----------------------
__global__ __launch_bounds__(256) void prep_kernel(const float* __restrict__ x,
                                                   bf16* __restrict__ Vb) {
    int idx = blockIdx.x * 256 + threadIdx.x;      // 32768*160 total
    int k   = idx % K1;
    int row = idx / K1;
    float val = 0.0f;
    if (k < 132) {
        int c, coord; float mean, inv_std;
        if (k < 80)       { c = (k >> 1);            coord = k & 1;        mean = 0.48f; inv_std = 1.0f / 0.25f; }
        else if (k < 122) { int kk = k - 80;  c = 40 + (kk >> 1); coord = kk & 1; mean = 0.50f; inv_std = 1.0f / 0.30f; }
        else              { int kk = k - 122; c = 61 + (kk >> 1); coord = kk & 1; mean = 0.50f; inv_std = 1.0f / 0.35f; }
        float v = x[((size_t)row * 66 + c) * 2 + coord];
        val = (v == 0.0f) ? 0.0f : (v - mean) * inv_std;
    }
    Vb[(size_t)row * K1 + k] = f2bf(val);
}

// ---------------- weight packing (TRANSPOSED to [N][K]) ---------------------
// W1cT: (1536,160) block-diag; W2wT: (512,1536) softmax-scaled;
// FC1T/FC2T: (512,512) transposed bf16 copies.
__global__ __launch_bounds__(256) void build_weights_kernel(
    const float* __restrict__ lips_w1, const float* __restrict__ lips_w2,
    const float* __restrict__ hand_w1, const float* __restrict__ hand_w2,
    const float* __restrict__ pose_w1, const float* __restrict__ pose_w2,
    const float* __restrict__ lw,
    const float* __restrict__ fc_w1,   const float* __restrict__ fc_w2,
    bf16* __restrict__ W1cT, bf16* __restrict__ W2wT,
    bf16* __restrict__ FC1T, bf16* __restrict__ FC2T) {
    const int N1 = H3 * K1;          // 245760
    const int N2 = DIM * H3;         // 786432
    const int N3 = DIM * DIM;        // 262144
    int idx = blockIdx.x * 256 + threadIdx.x;
    if (idx < N1) {
        int n = idx / K1, k = idx % K1;          // n: hidden col, k: feature
        int g = n / DIM, d = n % DIM;
        float v = 0.0f;
        if      (g == 0 && k < 80)              v = lips_w1[k * DIM + d];
        else if (g == 1 && k >= 80 && k < 122)  v = hand_w1[(k - 80) * DIM + d];
        else if (g == 2 && k >= 122 && k < 132) v = pose_w1[(k - 122) * DIM + d];
        W1cT[idx] = f2bf(v);
    } else if (idx < N1 + N2) {
        int j = idx - N1;
        int e = j / H3, r = j % H3;              // e: out col, r: hidden (g,d)
        int g = r / DIM, d = r % DIM;
        float w0 = lw[0], w1 = lw[1], w2 = lw[2];
        float mx = fmaxf(w0, fmaxf(w1, w2));
        float e0 = expf(w0 - mx), e1 = expf(w1 - mx), e2 = expf(w2 - mx);
        float inv_s = 1.0f / (e0 + e1 + e2);
        float a = ((g == 0) ? e0 : (g == 1) ? e1 : e2) * inv_s;
        const float* wp = (g == 0) ? lips_w2 : (g == 1) ? hand_w2 : pose_w2;
        W2wT[j] = f2bf(a * wp[d * DIM + e]);
    } else if (idx < N1 + N2 + N3) {
        int j = idx - N1 - N2;
        int e = j / DIM, d = j % DIM;
        FC1T[j] = f2bf(fc_w1[d * DIM + e]);
    } else if (idx < N1 + N2 + 2 * N3) {
        int j = idx - N1 - N2 - N3;
        int e = j / DIM, d = j % DIM;
        FC2T[j] = f2bf(fc_w2[d * DIM + e]);
    }
}

// ---------------- positional index -----------------------------------------
__global__ __launch_bounds__(256) void idx_kernel(const float* __restrict__ nefi,
                                                  int* __restrict__ norm_idx) {
    int b = blockIdx.x;                       // 64 blocks
    __shared__ float red[256];
    float m = -1e30f;
    for (int t = threadIdx.x; t < 512; t += 256) m = fmaxf(m, nefi[b * 512 + t]);
    red[threadIdx.x] = m;
    __syncthreads();
    for (int s = 128; s > 0; s >>= 1) {
        if (threadIdx.x < s) red[threadIdx.x] = fmaxf(red[threadIdx.x], red[threadIdx.x + s]);
        __syncthreads();
    }
    float mx = fmaxf(red[0], 1.0f);           // clip(max, 1.0)
    for (int t = threadIdx.x; t < 512; t += 256) {
        float v = nefi[b * 512 + t];
        norm_idx[b * 512 + t] = (v == -1.0f) ? 512 : (int)(v / mx * 512.0f);
    }
}

// ---------------- TDM descriptor + issue ------------------------------------
// 2D tile: tile_dim0 = 32 bf16 along K (64B rows), tile_dim1 = rows.
// pad_enable with interval=64B (code 3), amount=32B (code 7) -> LDS row
// stride becomes 96B = LDS_RS bf16 elements.
__device__ inline void tdm_issue(const bf16* gptr, unsigned lds_off,
                                 int Kelem, int tile_rows, int tensor_rows) {
    unsigned long long ga = (unsigned long long)(size_t)gptr;
    u32x4 g0;
    g0[0] = 1u;                                            // count=1, load, user
    g0[1] = lds_off;                                       // lds_addr (bytes)
    g0[2] = (unsigned)(ga & 0xFFFFFFFFu);                  // global_addr[31:0]
    g0[3] = (unsigned)((ga >> 32) & 0x01FFFFFFu) | (2u << 30);  // addr[56:32] | type=2
    i32x8 g1;
    // data_size=1 (2B) <<16 | pad_enable<<20 | pad_interval=3<<22 | pad_amount=7<<25
    g1[0] = (1 << 16) | (1 << 20) | (3 << 22) | (7 << 25);
    unsigned td0 = (unsigned)Kelem;        // tensor_dim0 (elements)
    unsigned td1 = (unsigned)tensor_rows;  // tensor_dim1 (rows)
    g1[1] = (int)((td0 & 0xFFFFu) << 16);                         // [63:48]=td0.lo
    g1[2] = (int)((td0 >> 16) | ((td1 & 0xFFFFu) << 16));         // td0.hi | td1.lo
    g1[3] = (int)((td1 >> 16) | ((unsigned)TILE_K << 16));        // td1.hi | tile_dim0
    g1[4] = (int)((unsigned)tile_rows);                           // tile_dim1 | tile_dim2=0
    g1[5] = (int)((unsigned)Kelem);                               // tensor_dim0_stride lo32
    g1[6] = 0;                                                    // stride0 hi16 | stride1 lo16
    g1[7] = 0;
    i32x4 z4;
    z4[0] = 0; z4[1] = 0; z4[2] = 0; z4[3] = 0;
    i32x8 z8;
    #pragma unroll
    for (int i = 0; i < 8; i++) z8[i] = 0;
    // 6-arg form (clang-23 / therock headers): groups 0..3 + aux + cpol
    __builtin_amdgcn_tensor_load_to_lds(g0, g1, z4, z4, z8, 0);
}

__device__ inline unsigned lds_off_of(const void* p) {
    return (unsigned)(size_t)p;   // LDS aperture: addr[31:0] = LDS byte offset
}

// ---------------- WMMA GEMM (B pre-transposed: BT is [N][K]) ----------------
// C[M,N] = A[M,K] @ BT[N,K]^T ; block tile 64x128, 8 waves, double-buffered
// TDM staging. MODE 0: gelu->bf16 ; 1: bf16 ; 2: f32 + pos_table add.
template<int MODE>
__global__ __launch_bounds__(256) void gemm_bf16_kernel(
    const bf16* __restrict__ A, const bf16* __restrict__ BT,
    void* __restrict__ Cout, int M, int N, int K,
    const int* __restrict__ norm_idx, const float* __restrict__ pos_table) {
    __shared__ alignas(16) bf16 Alds[2][TILE_M * LDS_RS];   // 2 x 6 KB
    __shared__ alignas(16) bf16 Blds[2][TILE_N * LDS_RS];   // 2 x 12 KB

    const int tid    = threadIdx.x;
    const int lane   = tid & 31;
    const int wid    = tid >> 5;
    const int lanelo = lane & 15;
    const int half   = lane >> 4;
    const int wm     = wid & 3;      // M tile within block
    const int wn     = wid >> 2;     // N quad within block
    const int m0     = blockIdx.y * TILE_M;
    const int n0     = blockIdx.x * TILE_N;

    v8f zero;
    #pragma unroll
    for (int i = 0; i < 8; i++) zero[i] = 0.0f;
    v8f acc[4] = {zero, zero, zero, zero};

    const int niter = K / TILE_K;
    const bf16* arow = A  + (size_t)m0 * K;
    const bf16* brow = BT + (size_t)n0 * K;

    // prologue: stage 0 into buffer 0
    if (wid == 0) {
        tdm_issue(arow, lds_off_of(&Alds[0][0]), K, TILE_M, M - m0);
        tdm_issue(brow, lds_off_of(&Blds[0][0]), K, TILE_N, N - n0);
    }

    for (int it = 0; it < niter; ++it) {
        const int cur = it & 1;
        if (wid == 0) {
            if (it + 1 < niter) {
                const int nxt = cur ^ 1;
                tdm_issue(arow + (size_t)(it + 1) * TILE_K,
                          lds_off_of(&Alds[nxt][0]), K, TILE_M, M - m0);
                tdm_issue(brow + (size_t)(it + 1) * TILE_K,
                          lds_off_of(&Blds[nxt][0]), K, TILE_N, N - n0);
                __builtin_amdgcn_s_wait_tensorcnt(2);   // stage `it` done (in-order)
            } else {
                __builtin_amdgcn_s_wait_tensorcnt(0);
            }
        }
        __syncthreads();   // buffer `cur` visible to all waves

        // ---- A fragment: row M=lanelo, K = half*8+{0..7}, 16+half*8+{0..7}
        v16bf afrag;
        {
            const bf16* base = &Alds[cur][(wm * 16 + lanelo) * LDS_RS];
            v8bf lo = *(const v8bf*)(base + half * 8);
            v8bf hi = *(const v8bf*)(base + 16 + half * 8);
            #pragma unroll
            for (int i = 0; i < 8; i++) { afrag[i] = lo[i]; afrag[8 + i] = hi[i]; }
        }
        // ---- 4 B fragments + WMMA ----
        #pragma unroll
        for (int j = 0; j < 4; j++) {
            const bf16* base = &Blds[cur][((wn * 4 + j) * 16 + lanelo) * LDS_RS];
            v8bf lo = *(const v8bf*)(base + half * 8);
            v8bf hi = *(const v8bf*)(base + 16 + half * 8);
            v16bf bfrag;
            #pragma unroll
            for (int i = 0; i < 8; i++) { bfrag[i] = lo[i]; bfrag[8 + i] = hi[i]; }
            acc[j] = __builtin_amdgcn_wmma_f32_16x16x32_bf16(
                false, afrag, false, bfrag, (short)0, acc[j], false, false);
        }
        __syncthreads();   // all reads of buffer `cur` done; safe to refill next iter
    }

    // ---- epilogue: C layout VGPR i -> row half*8+i, col lanelo ----
    #pragma unroll
    for (int j = 0; j < 4; j++) {
        const int col = n0 + (wn * 4 + j) * 16 + lanelo;
        #pragma unroll
        for (int i = 0; i < 8; i++) {
            const int row = m0 + wm * 16 + half * 8 + i;
            float v = acc[j][i];
            if (MODE == 0) {
                ((bf16*)Cout)[(size_t)row * N + col] = f2bf(gelu_exact(v));
            } else if (MODE == 1) {
                ((bf16*)Cout)[(size_t)row * N + col] = f2bf(v);
            } else {
                int pi = norm_idx[row];
                ((float*)Cout)[(size_t)row * N + col] = v + pos_table[(size_t)pi * DIM + col];
            }
        }
    }
}

// ---------------------------------------------------------------------------
extern "C" void kernel_launch(void* const* d_in, const int* in_sizes, int n_in,
                              void* d_out, int out_size, void* d_ws, size_t ws_size,
                              hipStream_t stream) {
    (void)in_sizes; (void)n_in; (void)out_size; (void)ws_size;
    const float* x       = (const float*)d_in[0];
    const float* nefi    = (const float*)d_in[1];
    const float* lips_w1 = (const float*)d_in[2];
    const float* lips_w2 = (const float*)d_in[3];
    const float* hand_w1 = (const float*)d_in[5];
    const float* hand_w2 = (const float*)d_in[6];
    const float* pose_w1 = (const float*)d_in[8];
    const float* pose_w2 = (const float*)d_in[9];
    const float* lw      = (const float*)d_in[11];
    const float* fc_w1   = (const float*)d_in[12];
    const float* fc_w2   = (const float*)d_in[13];
    const float* pos_tab = (const float*)d_in[14];

    // -------- workspace carve-up (256B aligned) --------
    char* ws = (char*)d_ws;
    size_t off = 0;
    auto carve = [&](size_t bytes) -> void* {
        void* p = ws + off;
        off += (bytes + 255) & ~((size_t)255);
        return p;
    };
    bf16* Vb   = (bf16*)carve((size_t)MROWS * K1  * sizeof(bf16));  // 10.5 MB
    bf16* W1cT = (bf16*)carve((size_t)H3    * K1  * sizeof(bf16));  // 0.5 MB
    bf16* W2wT = (bf16*)carve((size_t)DIM   * H3  * sizeof(bf16));  // 1.5 MB
    bf16* FC1T = (bf16*)carve((size_t)DIM   * DIM * sizeof(bf16));  // 0.5 MB
    bf16* FC2T = (bf16*)carve((size_t)DIM   * DIM * sizeof(bf16));  // 0.5 MB
    bf16* H    = (bf16*)carve((size_t)MROWS * H3  * sizeof(bf16));  // 100 MB
    bf16* EMB  = (bf16*)carve((size_t)MROWS * DIM * sizeof(bf16));  // 33.5 MB
    bf16* H1   = (bf16*)carve((size_t)MROWS * DIM * sizeof(bf16));  // 33.5 MB
    int*  NID  = (int*) carve((size_t)MROWS * sizeof(int));         // 128 KB

    // 1) pack weights (transposed to [N][K])
    {
        int total = H3 * K1 + DIM * H3 + 2 * DIM * DIM;  // 1,556,480
        build_weights_kernel<<<(total + 255) / 256, 256, 0, stream>>>(
            lips_w1, lips_w2, hand_w1, hand_w2, pose_w1, pose_w2, lw,
            fc_w1, fc_w2, W1cT, W2wT, FC1T, FC2T);
    }
    // 2) normalize + pack features
    prep_kernel<<<(MROWS * K1) / 256, 256, 0, stream>>>(x, Vb);
    // 3) positional indices
    idx_kernel<<<64, 256, 0, stream>>>(nefi, NID);

    // 4) H = gelu(Vb @ W1c)                 [32768 x 1536], K=160
    gemm_bf16_kernel<0><<<dim3(H3 / TILE_N, MROWS / TILE_M), 256, 0, stream>>>(
        Vb, W1cT, H, MROWS, H3, K1, nullptr, nullptr);
    // 5) EMB = H @ W2w (softmax-weighted group sum)   [32768 x 512], K=1536
    gemm_bf16_kernel<1><<<dim3(DIM / TILE_N, MROWS / TILE_M), 256, 0, stream>>>(
        H, W2wT, EMB, MROWS, DIM, H3, nullptr, nullptr);
    // 6) H1 = gelu(EMB @ fc_w1)             [32768 x 512], K=512
    gemm_bf16_kernel<0><<<dim3(DIM / TILE_N, MROWS / TILE_M), 256, 0, stream>>>(
        EMB, FC1T, H1, MROWS, DIM, DIM, nullptr, nullptr);
    // 7) out = H1 @ fc_w2 + pos_table[norm_idx]   (f32 out)
    gemm_bf16_kernel<2><<<dim3(DIM / TILE_N, MROWS / TILE_M), 256, 0, stream>>>(
        H1, FC2T, d_out, MROWS, DIM, DIM, NID, pos_tab);
}
